// MoELanguageZone_52415780880471
// MI455X (gfx1250) — matrix operand order repaired and checked by reading, not compiled, not run on hardware
//
#include <hip/hip_runtime.h>

typedef __attribute__((ext_vector_type(16))) __bf16 v16bf;
typedef __attribute__((ext_vector_type(8)))  float  v8f;

#define TOK_N 4096
#define DIM_D 1024
#define DIM_H 1024
#define DIM_M 64
#define NUM_E 8
#define DIM_HE 512
#define VOCAB 32000

// ---------------------------------------------------------------------------
// bf16 WMMA GEMM:  out = act(A[M,K] @ B[K,N] + bias), B given pre-TRANSPOSED
// as BT[N,K].  Block: 128 x BNT, 256 threads = 8 waves (4 row x 2 col groups),
// wave tile 32 x (BNT/2).  Double-buffered LDS, one barrier per k-step.
// ---------------------------------------------------------------------------
#define BM 128
#define BK 32
#define LDSS 56   // LDS row stride in bf16 (112B): 16B-aligned, conflict-free frags

template<int ACT, int OMODE, int BNT>  // ACT: 0 none,1 sigmoid,2 relu; OMODE bit0=f32,bit1=bf16
__global__ __launch_bounds__(256) void gemm_kernel(
    const __bf16* __restrict__ A,  long sA,
    const __bf16* __restrict__ BT, long sB,
    const float*  __restrict__ bias, long sBias,
    float*  __restrict__ outF, long sOutF,
    __bf16* __restrict__ outH, long sOutH,
    int N, int K)
{
    constexpr int CT = BNT / 32;            // col tiles per wave (2 or 4)
    __shared__ __bf16 As[2][BM  * LDSS];
    __shared__ __bf16 Bs[2][BNT * LDSS];

    const int batch = blockIdx.z;
    A  += (long)batch * sA;
    BT += (long)batch * sB;

    const int rowBase = blockIdx.x * BM;
    const int colBase = blockIdx.y * BNT;
    const int t    = threadIdx.x;
    const int wid  = t >> 5;
    const int lane = t & 31;
    const int waveRow = (wid >> 1) * 32;           // 0,32,64,96
    const int waveCol = (wid & 1)  * (BNT / 2);    // 0, BNT/2

    // staging indices (A tile: 128x32, each thread 32B; B tile: BNTx32)
    const int ar = t >> 1, ah = (t & 1) * 16;
    const __bf16* aSrc0 = A + (long)(rowBase + ar) * K + ah;
    const int aOff = ar * LDSS + ah;
    int br, bh;
    if (BNT == 128) { br = t >> 1; bh = (t & 1) * 16; }
    else            { br = t >> 2; bh = (t & 3) * 8;  }
    const __bf16* bSrc0 = BT + (long)(colBase + br) * K + bh;
    const int bOff = br * LDSS + bh;

    // fragment read offsets (ISA 7.12.2): A lane: row l&15, K chunks (l>>4)*8, +16
    const int aFrow = waveRow + (lane & 15);
    const int akb   = (lane >> 4) * 8;
    const int bFcol = waveCol + (lane & 15);
    const int bkb   = (lane >> 4) * 8;             // B: K chunks (l>>4)*8, +16 (same as A)

    v8f acc[2][CT] = {};
    uint4 ra0, ra1, rb0, rb1;

    // prologue: stage k-step 0 into buffer 0
    ra0 = *(const uint4*)(aSrc0);
    ra1 = *(const uint4*)(aSrc0 + 8);
    rb0 = *(const uint4*)(bSrc0);
    if (BNT == 128) rb1 = *(const uint4*)(bSrc0 + 8);
    *(uint4*)(&As[0][aOff])     = ra0;
    *(uint4*)(&As[0][aOff + 8]) = ra1;
    *(uint4*)(&Bs[0][bOff])     = rb0;
    if (BNT == 128) *(uint4*)(&Bs[0][bOff + 8]) = rb1;
    __syncthreads();

    const int nIter = K / BK;
    for (int it = 0; it < nIter; ++it) {
        const int cur = it & 1;
        if (it + 1 < nIter) {                       // issue next tile's loads
            const int kk = (it + 1) * BK;
            ra0 = *(const uint4*)(aSrc0 + kk);
            ra1 = *(const uint4*)(aSrc0 + kk + 8);
            rb0 = *(const uint4*)(bSrc0 + kk);
            if (BNT == 128) rb1 = *(const uint4*)(bSrc0 + kk + 8);
        }
        // fragments + WMMA on current buffer
        v16bf afrag[2], bfrag[CT];
        #pragma unroll
        for (int i = 0; i < 2; ++i) {
            const __bf16* p = &As[cur][(aFrow + i * 16) * LDSS];
            union { v16bf v; uint4 q[2]; } u;
            u.q[0] = *(const uint4*)(p + akb);
            u.q[1] = *(const uint4*)(p + akb + 16);
            afrag[i] = u.v;
        }
        #pragma unroll
        for (int j = 0; j < CT; ++j) {
            const __bf16* p = &Bs[cur][(bFcol + j * 16) * LDSS + bkb];
            union { v16bf v; uint4 q[2]; } u;
            u.q[0] = *(const uint4*)(p);
            u.q[1] = *(const uint4*)(p + 16);
            bfrag[j] = u.v;
        }
        #pragma unroll
        for (int i = 0; i < 2; ++i)
            #pragma unroll
            for (int j = 0; j < CT; ++j)
                acc[i][j] = __builtin_amdgcn_wmma_f32_16x16x32_bf16(
                    false, afrag[i], false, bfrag[j], (short)0, acc[i][j], false, false);

        if (it + 1 < nIter) {                       // write next buffer
            *(uint4*)(&As[cur ^ 1][aOff])     = ra0;
            *(uint4*)(&As[cur ^ 1][aOff + 8]) = ra1;
            *(uint4*)(&Bs[cur ^ 1][bOff])     = rb0;
            if (BNT == 128) *(uint4*)(&Bs[cur ^ 1][bOff + 8]) = rb1;
        }
        __syncthreads();
    }

    // epilogue: lane l -> col l&15; VGPR v -> row v + 8*(l>>4)
    const int crow0 = rowBase + waveRow + ((lane >> 4) * 8);
    const int ccol0 = colBase + waveCol + (lane & 15);
    float bv[CT];
    #pragma unroll
    for (int j = 0; j < CT; ++j)
        bv[j] = bias ? bias[(long)batch * sBias + ccol0 + j * 16] : 0.0f;

    float*  pF = (OMODE & 1) ? outF + (long)batch * sOutF + (long)crow0 * N + ccol0 : nullptr;
    __bf16* pH = (OMODE & 2) ? outH + (long)batch * sOutH + (long)crow0 * N + ccol0 : nullptr;
    #pragma unroll
    for (int i = 0; i < 2; ++i) {
        #pragma unroll
        for (int j = 0; j < CT; ++j) {
            #pragma unroll
            for (int v = 0; v < 8; ++v) {
                float x = acc[i][j][v] + bv[j];
                if (ACT == 1)      x = 1.0f / (1.0f + expf(-x));
                else if (ACT == 2) x = x > 0.0f ? x : 0.0f;
                const int ofs = (i * 16 + v) * N + j * 16;
                if (OMODE & 1) pF[ofs] = x;
                if (OMODE & 2) pH[ofs] = (__bf16)x;
            }
        }
    }
}

template<int ACT, int OMODE, int BNT>
static void run_gemm(const __bf16* A, long sA, const __bf16* BT, long sB,
                     const float* bias, long sBias,
                     float* outF, long sOutF, __bf16* outH, long sOutH,
                     int M, int N, int K, int batch, hipStream_t st) {
    gemm_kernel<ACT, OMODE, BNT><<<dim3(M / BM, N / BNT, batch), 256, 0, st>>>(
        A, sA, BT, sB, bias, sBias, outF, sOutF, outH, sOutH, N, K);
}

// ---------------------------------------------------------------------------
// Tiled transpose + f32->bf16:  src[K,N] (f32)  ->  dst[N,K] (bf16)
// ---------------------------------------------------------------------------
__global__ __launch_bounds__(256) void transpose_bf16_kernel(
    const float* __restrict__ src, __bf16* __restrict__ dst, int K, int N) {
    __shared__ float tile[32][33];
    const long bofs = (long)blockIdx.z * K * N;
    src += bofs; dst += bofs;
    const int k0 = blockIdx.x * 32, n0 = blockIdx.y * 32;
    const int tx = threadIdx.x & 31, ty = threadIdx.x >> 5;  // ty 0..7
    #pragma unroll
    for (int i = 0; i < 4; ++i)
        tile[ty + i * 8][tx] = src[(long)(k0 + ty + i * 8) * N + n0 + tx];
    __syncthreads();
    #pragma unroll
    for (int i = 0; i < 4; ++i)
        dst[(long)(n0 + ty + i * 8) * K + k0 + tx] = (__bf16)tile[tx][ty + i * 8];
}

__global__ __launch_bounds__(256) void gather_embed_kernel(
    const int* __restrict__ ids, const float* __restrict__ emb,
    __bf16* __restrict__ x) {
    const int n = blockIdx.x;
    const long row = (long)ids[n];
    const int t = threadIdx.x;  // 256 threads * 4 floats = 1024
    const float4 v = *(const float4*)(emb + row * DIM_D + t * 4);
    union { __bf16 h[4]; unsigned long long u; } o;
    o.h[0] = (__bf16)v.x; o.h[1] = (__bf16)v.y; o.h[2] = (__bf16)v.z; o.h[3] = (__bf16)v.w;
    *(unsigned long long*)(x + (long)n * DIM_D + t * 4) = o.u;
}

// router: gate_h = tanh(cont@rW1+rb1); logits = gate_h@rW2+rb2; softmax; top-2
__global__ void router_kernel(const float* __restrict__ cont,
                              const float* __restrict__ rW1, const float* __restrict__ rb1,
                              const float* __restrict__ rW2, const float* __restrict__ rb2,
                              float* __restrict__ w_full, float* __restrict__ probs_out) {
    const int n = blockIdx.x;
    const int j = threadIdx.x;  // 0..63
    __shared__ float sc[64], sg[64], sl[8];
    sc[j] = cont[(long)n * 64 + j];
    __syncthreads();
    float a = rb1[j];
    for (int k = 0; k < 64; ++k) a += sc[k] * rW1[k * 64 + j];
    sg[j] = tanhf(a);
    __syncthreads();
    if (j < 8) {
        float b = rb2[j];
        for (int k = 0; k < 64; ++k) b += sg[k] * rW2[k * 8 + j];
        sl[j] = b;
    }
    __syncthreads();
    if (j == 0) {
        float mx = sl[0];
        for (int e = 1; e < 8; ++e) mx = fmaxf(mx, sl[e]);
        float p[8], s = 0.0f;
        for (int e = 0; e < 8; ++e) { p[e] = expf(sl[e] - mx); s += p[e]; }
        const float inv = 1.0f / s;
        for (int e = 0; e < 8; ++e) { p[e] *= inv; probs_out[(long)n * 8 + e] = p[e]; }
        int i0 = 0; float v0 = -1.0f;
        for (int e = 0; e < 8; ++e) if (p[e] > v0) { v0 = p[e]; i0 = e; }
        int i1 = 0; float v1 = -1.0f;
        for (int e = 0; e < 8; ++e) if (e != i0 && p[e] > v1) { v1 = p[e]; i1 = e; }
        const float tot = v0 + v1;
        for (int e = 0; e < 8; ++e) w_full[(long)n * 8 + e] = 0.0f;
        w_full[(long)n * 8 + i0] = v0 / tot;
        w_full[(long)n * 8 + i1] = v1 / tot;
    }
}

__global__ void combine_kernel(const float* __restrict__ eo,
                               const float* __restrict__ w_full,
                               __bf16* __restrict__ outH) {
    const int idx = blockIdx.x * blockDim.x + threadIdx.x;
    if (idx >= TOK_N * DIM_M) return;
    const int n = idx >> 6;
    float s = 0.0f;
    #pragma unroll
    for (int e = 0; e < NUM_E; ++e)
        s += w_full[(long)n * 8 + e] * eo[(long)e * TOK_N * DIM_M + idx];
    outH[idx] = (__bf16)s;
}

// ---------------------------------------------------------------------------
extern "C" void kernel_launch(void* const* d_in, const int* in_sizes, int n_in,
                              void* d_out, int out_size, void* d_ws, size_t ws_size,
                              hipStream_t stream) {
    (void)in_sizes; (void)n_in; (void)out_size; (void)ws_size;
    const int*   ids   = (const int*)d_in[0];
    const float* emb   = (const float*)d_in[1];
    const float* enc_W = (const float*)d_in[2];
    const float* enc_b = (const float*)d_in[3];
    const float* s2c_W = (const float*)d_in[4];
    const float* s2c_b = (const float*)d_in[5];
    const float* rW1   = (const float*)d_in[6];
    const float* rb1   = (const float*)d_in[7];
    const float* rW2   = (const float*)d_in[8];
    const float* rb2   = (const float*)d_in[9];
    const float* eW1   = (const float*)d_in[10];
    const float* eb1   = (const float*)d_in[11];
    const float* eW2   = (const float*)d_in[12];
    const float* eb2   = (const float*)d_in[13];
    const float* c2s_W = (const float*)d_in[14];
    const float* c2s_b = (const float*)d_in[15];
    const float* dec_W = (const float*)d_in[16];
    const float* dec_b = (const float*)d_in[17];
    const float* out_W = (const float*)d_in[18];
    const float* out_b = (const float*)d_in[19];

    float* logits = (float*)d_out;                 // (4096, 32000)
    float* probs  = logits + (long)TOK_N * VOCAB;  // (4096, 8)

    // ---- workspace carve-out (256B aligned) ----
    char* ws = (char*)d_ws;
    size_t off = 0;
    auto carve = [&](size_t bytes) { void* p = ws + off; off = (off + bytes + 255) & ~(size_t)255; return p; };
    __bf16* x_bf    = (__bf16*)carve((size_t)TOK_N * DIM_D * 2);          // reused: spikes_moe
    __bf16* encWT   = (__bf16*)carve((size_t)DIM_D * DIM_H * 2);          // [H,D]
    __bf16* senc_bf = (__bf16*)carve((size_t)TOK_N * DIM_H * 2);          // reused: decoded
    __bf16* s2cWT   = (__bf16*)carve((size_t)DIM_H * DIM_M * 2);          // [M,H]
    float*  cont_f  = (float*) carve((size_t)TOK_N * DIM_M * 4);
    __bf16* cont_bf = (__bf16*)carve((size_t)TOK_N * DIM_M * 2);
    float*  w_full  = (float*) carve((size_t)TOK_N * NUM_E * 4);
    __bf16* eW1T    = (__bf16*)carve((size_t)NUM_E * DIM_M * DIM_HE * 2); // [E][HE,M]
    __bf16* eW2T    = (__bf16*)carve((size_t)NUM_E * DIM_HE * DIM_M * 2); // [E][M,HE]
    __bf16* h1_bf   = (__bf16*)carve((size_t)NUM_E * TOK_N * DIM_HE * 2);
    float*  eo_f    = (float*) carve((size_t)NUM_E * TOK_N * DIM_M * 4);
    __bf16* eout_bf = (__bf16*)carve((size_t)TOK_N * DIM_M * 2);
    __bf16* c2sWT   = (__bf16*)carve((size_t)DIM_M * DIM_H * 2);          // [H,M]
    __bf16* decWT   = (__bf16*)carve((size_t)DIM_H * DIM_D * 2);          // [D,H]
    __bf16* outWT   = (__bf16*)carve((size_t)DIM_D * VOCAB * 2);          // [V,D] 64MB -> L2-resident
    __bf16* smoe_bf = x_bf;
    __bf16* dec_bf  = senc_bf;

    // ---- one-time weight transpose+convert (coalesced 32x32 LDS tiles) ----
    transpose_bf16_kernel<<<dim3(DIM_D/32, DIM_H/32, 1), 256, 0, stream>>>(enc_W, encWT, DIM_D, DIM_H);
    transpose_bf16_kernel<<<dim3(DIM_H/32, DIM_M/32, 1), 256, 0, stream>>>(s2c_W, s2cWT, DIM_H, DIM_M);
    transpose_bf16_kernel<<<dim3(DIM_M/32, DIM_HE/32, NUM_E), 256, 0, stream>>>(eW1, eW1T, DIM_M, DIM_HE);
    transpose_bf16_kernel<<<dim3(DIM_HE/32, DIM_M/32, NUM_E), 256, 0, stream>>>(eW2, eW2T, DIM_HE, DIM_M);
    transpose_bf16_kernel<<<dim3(DIM_M/32, DIM_H/32, 1), 256, 0, stream>>>(c2s_W, c2sWT, DIM_M, DIM_H);
    transpose_bf16_kernel<<<dim3(DIM_H/32, DIM_D/32, 1), 256, 0, stream>>>(dec_W, decWT, DIM_H, DIM_D);
    transpose_bf16_kernel<<<dim3(DIM_D/32, VOCAB/32, 1), 256, 0, stream>>>(out_W, outWT, DIM_D, VOCAB);

    // 1) embed gather -> bf16
    gather_embed_kernel<<<dim3(TOK_N), dim3(256), 0, stream>>>(ids, emb, x_bf);

    // 2) spikes_enc = sigmoid(x @ enc_W + enc_b)            (4096x1024, K=1024)
    run_gemm<1, 2, 128>(x_bf, 0, encWT, 0, enc_b, 0,
                        nullptr, 0, senc_bf, 0, TOK_N, DIM_H, DIM_D, 1, stream);

    // 3) cont = spikes_enc @ s2c_W + s2c_b                  (4096x64, K=1024)
    run_gemm<0, 3, 64>(senc_bf, 0, s2cWT, 0, s2c_b, 0,
                       cont_f, 0, cont_bf, 0, TOK_N, DIM_M, DIM_H, 1, stream);

    // 4) router -> w_full, probs
    router_kernel<<<dim3(TOK_N), dim3(64), 0, stream>>>(cont_f, rW1, rb1, rW2, rb2, w_full, probs);

    // 5) h1[e] = relu(cont @ eW1[e] + eb1[e])               (8 x 4096x512, K=64)
    run_gemm<2, 2, 128>(cont_bf, 0, eW1T, (long)DIM_M * DIM_HE, eb1, DIM_HE,
                        nullptr, 0, h1_bf, (long)TOK_N * DIM_HE,
                        TOK_N, DIM_HE, DIM_M, NUM_E, stream);

    // 6) eo[e] = h1[e] @ eW2[e] + eb2[e]                    (8 x 4096x64, K=512)
    run_gemm<0, 1, 64>(h1_bf, (long)TOK_N * DIM_HE, eW2T, (long)DIM_HE * DIM_M, eb2, DIM_M,
                       eo_f, (long)TOK_N * DIM_M, nullptr, 0,
                       TOK_N, DIM_M, DIM_HE, NUM_E, stream);

    // 7) weighted combine over experts
    combine_kernel<<<dim3((TOK_N * DIM_M + 255) / 256), dim3(256), 0, stream>>>(eo_f, w_full, eout_bf);

    // 8) spikes_moe = sigmoid(expert_out @ c2s_W + c2s_b)   (4096x1024, K=64)
    run_gemm<1, 2, 128>(eout_bf, 0, c2sWT, 0, c2s_b, 0,
                        nullptr, 0, smoe_bf, 0, TOK_N, DIM_H, DIM_M, 1, stream);

    // 9) decoded = sigmoid(spikes_moe @ dec_W + dec_b)      (4096x1024, K=1024)
    run_gemm<1, 2, 128>(smoe_bf, 0, decWT, 0, dec_b, 0,
                        nullptr, 0, dec_bf, 0, TOK_N, DIM_D, DIM_H, 1, stream);

    // 10) logits = decoded @ out_W + out_b                  (4096x32000, K=1024)
    run_gemm<0, 1, 128>(dec_bf, 0, outWT, 0, out_b, 0,
                        logits, 0, nullptr, 0, TOK_N, VOCAB, DIM_D, 1, stream);
}
